// GNN_47940424958091
// MI455X (gfx1250) — matrix operand order repaired and checked by reading, not compiled, not run on hardware
//
#include <hip/hip_runtime.h>
#include <hip/hip_bf16.h>

// ---------------------------------------------------------------------------
// GNN message passing on MI455X (gfx1250): bf16 WMMA 16x16x32, fp32 accum.
// Weights pre-swizzled into B-fragment order (b128 loads only).
// Register blocking: per kt-step, A-fragments loaded once and reused across
// all n-tiles owned by the wave; B-fragments reused across 2 M-tiles.
// Edge gathers use GLOBAL_LOAD_ASYNC_TO_LDS_B128 (ASYNCcnt) - no VGPR staging.
// ---------------------------------------------------------------------------

typedef __bf16 bf16;
typedef __attribute__((ext_vector_type(16))) __bf16 v16bf;
typedef __attribute__((ext_vector_type(8)))  __bf16 v8bf;
typedef __attribute__((ext_vector_type(4)))  __bf16 v4bf;
typedef __attribute__((ext_vector_type(8)))  float  v8f;

#define NNODE  20000
#define NEDGE  320000
#define NEDGE1 160000
#define TITER  3

static __device__ __forceinline__ v8f wmma_bf16(v16bf a, v16bf b, v8f c) {
  return __builtin_amdgcn_wmma_f32_16x16x32_bf16(
      /*neg_a=*/false, a, /*neg_b=*/false, b,
      /*c_mod=*/(short)0, c, /*reuse_a=*/false, /*reuse_b=*/false);
}

static __device__ __forceinline__ v16bf cat8(v8bf lo, v8bf hi) {
  return __builtin_shufflevector(lo, hi, 0, 1, 2, 3, 4, 5, 6, 7,
                                         8, 9, 10, 11, 12, 13, 14, 15);
}

// Async DMA: 16 bytes global -> LDS, tracked by ASYNCcnt (no VGPR staging).
// LDS operand = wave-relative low 32 bits of the flat shared-pointer address
// (ISA 10.2: LDS aperture address = addr[31:0]).
static __device__ __forceinline__ void async_gather_b128(void* ldsDst, const void* gsrc) {
  unsigned lds_addr = (unsigned)(size_t)ldsDst;
  asm volatile("global_load_async_to_lds_b128 %0, %1, off"
               :: "v"(lds_addr), "v"(gsrc)
               : "memory");
}

static __device__ __forceinline__ void wait_async_zero() {
  asm volatile("s_wait_asynccnt 0" ::: "memory");
}

// Register-blocked GEMM: NTW n-tiles x MT m-tiles accumulated in registers.
// A: (MT*16 x K) row-major bf16 (stride sa, 16B-aligned rows), usually LDS.
// Bp: pre-swizzled fragment-major weights; tile (kt,nt) = 512 contiguous bf16.
// Per kt: MT A-fragment loads, NTW B-fragment loads, NTW*MT WMMAs.
template<int K, int NT, int NTW, int MT>
static __device__ __forceinline__ void gemm_block(const bf16* A, int sa,
                                                  const bf16* Bp, int ntBase,
                                                  v8f (&c)[NTW][MT]) {
  const int lane = threadIdx.x & 31;
  const int m    = lane & 15;
  const int half = lane >> 4;
  #pragma unroll
  for (int kt = 0; kt < K / 32; ++kt) {
    v16bf af[MT];
    #pragma unroll
    for (int mt = 0; mt < MT; ++mt) {
      const bf16* ap = A + (mt * 16 + m) * sa + kt * 32 + half * 16;
      af[mt] = cat8(*(const v8bf*)(ap), *(const v8bf*)(ap + 8));
    }
    #pragma unroll
    for (int ntw = 0; ntw < NTW; ++ntw) {
      const bf16* bp = Bp + ((size_t)(kt * NT + ntBase + ntw) * 32 + lane) * 16;
      v16bf bfrag = cat8(*(const v8bf*)(bp), *(const v8bf*)(bp + 8));
      #pragma unroll
      for (int mt = 0; mt < MT; ++mt)
        c[ntw][mt] = wmma_bf16(af[mt], bfrag, c[ntw][mt]);
    }
  }
}

static __device__ __forceinline__ v8f bias_frag(const float* b, int n0) {
  const float bv = b[n0 + (threadIdx.x & 15)];
  v8f c;
  #pragma unroll
  for (int j = 0; j < 8; ++j) c[j] = bv;
  return c;
}

static __device__ __forceinline__ void store_relu_bf16(bf16* out, int stride, v8f c) {
  const int lane = threadIdx.x & 31;
  const int n = lane & 15;
  const int half = lane >> 4;
  #pragma unroll
  for (int j = 0; j < 8; ++j) {
    float x = c[j];
    out[(j + 8 * half) * stride + n] = (bf16)(x > 0.f ? x : 0.f);
  }
}

// Layers 1+2 of (IN -> 256 -> 128) for a 32-row tile; 128 threads = 4 waves.
template<int IN>
static __device__ __forceinline__ void mlp_layers12(const bf16* X, bf16* H1s, bf16* H2s,
                                                    const bf16* W1p, const float* b1,
                                                    const bf16* W2p, const float* b2) {
  const int wave = threadIdx.x >> 5;
  {
    v8f c[4][2];
    #pragma unroll
    for (int t = 0; t < 4; ++t) {
      v8f cb = bias_frag(b1, (wave * 4 + t) * 16);
      c[t][0] = cb; c[t][1] = cb;
    }
    gemm_block<IN, 16, 4, 2>(X, IN, W1p, wave * 4, c);
    #pragma unroll
    for (int t = 0; t < 4; ++t) {
      store_relu_bf16(H1s + (wave * 4 + t) * 16, 256, c[t][0]);
      store_relu_bf16(H1s + 16 * 256 + (wave * 4 + t) * 16, 256, c[t][1]);
    }
  }
  __syncthreads();
  {
    v8f c[2][2];
    #pragma unroll
    for (int t = 0; t < 2; ++t) {
      v8f cb = bias_frag(b2, (wave * 2 + t) * 16);
      c[t][0] = cb; c[t][1] = cb;
    }
    gemm_block<256, 8, 2, 2>(H1s, 256, W2p, wave * 2, c);
    #pragma unroll
    for (int t = 0; t < 2; ++t) {
      store_relu_bf16(H2s + (wave * 2 + t) * 16, 128, c[t][0]);
      store_relu_bf16(H2s + 16 * 128 + (wave * 2 + t) * 16, 128, c[t][1]);
    }
  }
  __syncthreads();
}

// ---------------------------------------------------------------------------
// Kernels
// ---------------------------------------------------------------------------

// Repack row-major f32 weight (Kdim x Ndim) into bf16 WMMA-B fragment order:
// P[((kt*NT+nt)*32 + lane)*16 + i] = W[(kt*32 + (lane>>4)*16 + i)*Ndim + nt*16 + (lane&15)]
__global__ void repack_weight_kernel(const float* __restrict__ W, bf16* __restrict__ P,
                                     int Kdim, int Ndim) {
  int idx = blockIdx.x * 256 + threadIdx.x;
  if (idx >= Kdim * Ndim) return;
  const int i    = idx & 15;
  const int lane = (idx >> 4) & 31;
  const int tile = idx >> 9;
  const int NT = Ndim >> 4;
  const int kt = tile / NT;
  const int nt = tile - kt * NT;
  const int k = kt * 32 + (lane >> 4) * 16 + i;
  const int n = nt * 16 + (lane & 15);
  P[idx] = (bf16)W[k * Ndim + n];
}

// mlp_e evaluated only at the two distinct labels (1.0 and 0.0).
__global__ __launch_bounds__(256) void he_kernel(const float* W1, const float* b1,
                                                 const float* W2, const float* b2,
                                                 const float* W3, const float* b3,
                                                 bf16* he1, bf16* he0) {
  __shared__ float h1[256];
  __shared__ float h2[128];
  const int t = threadIdx.x;
  for (int lab = 0; lab < 2; ++lab) {
    const float l = (lab == 0) ? 1.f : 0.f;
    h1[t] = fmaxf(l * W1[t] + b1[t], 0.f);
    __syncthreads();
    if (t < 128) {
      float acc = b2[t];
      for (int j = 0; j < 256; ++j) acc += h1[j] * W2[j * 128 + t];
      h2[t] = fmaxf(acc, 0.f);
    }
    __syncthreads();
    if (t < 128) {
      float acc = b3[t];
      for (int j = 0; j < 128; ++j) acc += h2[j] * W3[j * 128 + t];
      (lab == 0 ? he1 : he0)[t] = (bf16)acc;
    }
    __syncthreads();
  }
}

// h = mlp_v(x_v): 128 -> 256 -> 128 -> 128, one 32-node tile per block.
__global__ __launch_bounds__(128) void node_mlp_kernel(const float* __restrict__ xv,
    const bf16* W1p, const float* b1, const bf16* W2p, const float* b2,
    const bf16* W3p, const float* b3, float* __restrict__ hf, bf16* __restrict__ hb) {
  __shared__ bf16 X[32 * 128];
  __shared__ bf16 H1s[32 * 256];
  __shared__ bf16 H2s[32 * 128];
  const int t = threadIdx.x;
  const int node0 = blockIdx.x * 32;
  #pragma unroll
  for (int q = t; q < 1024; q += 128) {
    const int row = q >> 5;
    const int off = (q & 31) * 4;
    float4 xq = ((const float4*)xv)[(size_t)(node0 + row) * 32 + (q & 31)];
    v4bf o; o[0] = (bf16)xq.x; o[1] = (bf16)xq.y; o[2] = (bf16)xq.z; o[3] = (bf16)xq.w;
    *(v4bf*)(&X[row * 128 + off]) = o;
  }
  __syncthreads();
  mlp_layers12<128>(X, H1s, H2s, W1p, b1, W2p, b2);
  const int lane = t & 31, wave = t >> 5;
  const int n = lane & 15, half = lane >> 4;
  v8f c[2][2];
  #pragma unroll
  for (int tt = 0; tt < 2; ++tt) {
    v8f cb = bias_frag(b3, (wave * 2 + tt) * 16);
    c[tt][0] = cb; c[tt][1] = cb;
  }
  gemm_block<128, 8, 2, 2>(H2s, 128, W3p, wave * 2, c);
  #pragma unroll
  for (int tt = 0; tt < 2; ++tt) {
    const int n0 = (wave * 2 + tt) * 16;
    #pragma unroll
    for (int mt = 0; mt < 2; ++mt) {
      #pragma unroll
      for (int j = 0; j < 8; ++j) {
        const int row = node0 + mt * 16 + j + 8 * half;
        const float x = c[tt][mt][j];
        hf[(size_t)row * 128 + n0 + n] = x;
        hb[(size_t)row * 128 + n0 + n] = (bf16)x;
      }
    }
  }
}

__global__ void count_kernel(const int* __restrict__ dst, float* c1, float* c0) {
  int e = blockIdx.x * 256 + threadIdx.x;
  if (e < NEDGE1)      atomicAdd(&c1[dst[e]], 1.f);
  else if (e < NEDGE)  atomicAdd(&c0[dst[e]], 1.f);
}

// Edge MLP: async-gather [h[s] | h[d] | h_e] (32x384) straight into LDS,
// 384->256->128->128, scatter-add into per-destination sum accumulator.
__global__ __launch_bounds__(128) void edge_mlp_kernel(const bf16* __restrict__ hb,
    const int* __restrict__ srcI, const int* __restrict__ dstI, const bf16* __restrict__ he,
    const bf16* W1p, const float* b1, const bf16* W2p, const float* b2,
    const bf16* W3p, const float* b3, float* __restrict__ a_sum) {
  __shared__ bf16 X[32 * 384];
  __shared__ bf16 H1s[32 * 256];
  __shared__ bf16 H2s[32 * 128];
  __shared__ int sidx[32];
  __shared__ int didx[32];
  const int t = threadIdx.x;
  const int e0 = blockIdx.x * 32;
  if (t < 32) { sidx[t] = srcI[e0 + t]; didx[t] = dstI[e0 + t]; }
  __syncthreads();
  // Gathered segments: 2 x (32 rows x 128 bf16) = 1024 async b128 DMAs.
  #pragma unroll
  for (int q = t; q < 1024; q += 128) {
    const int seg = q >> 9;
    const int row = (q >> 4) & 31;
    const int off = (q & 15) * 8;
    const int node = seg ? didx[row] : sidx[row];
    async_gather_b128(&X[row * 384 + seg * 128 + off],
                      hb + (size_t)node * 128 + off);
  }
  // Broadcast h_e segment: 512 async b128 DMAs (same 16 chunks per row).
  #pragma unroll
  for (int q = t; q < 512; q += 128) {
    const int row = q >> 4;
    const int off = (q & 15) * 8;
    async_gather_b128(&X[row * 384 + 256 + off], he + off);
  }
  wait_async_zero();
  __syncthreads();
  mlp_layers12<384>(X, H1s, H2s, W1p, b1, W2p, b2);
  const int lane = t & 31, wave = t >> 5;
  const int n = lane & 15, half = lane >> 4;
  v8f c[2][2];
  #pragma unroll
  for (int tt = 0; tt < 2; ++tt) {
    v8f cb = bias_frag(b3, (wave * 2 + tt) * 16);
    c[tt][0] = cb; c[tt][1] = cb;
  }
  gemm_block<128, 8, 2, 2>(H2s, 128, W3p, wave * 2, c);
  #pragma unroll
  for (int tt = 0; tt < 2; ++tt) {
    const int n0 = (wave * 2 + tt) * 16;
    #pragma unroll
    for (int mt = 0; mt < 2; ++mt) {
      #pragma unroll
      for (int j = 0; j < 8; ++j) {
        const int row = mt * 16 + j + 8 * half;
        atomicAdd(&a_sum[(size_t)didx[row] * 128 + n0 + n], c[tt][mt][j]);
      }
    }
  }
}

// h += mlp_aggr([h | a1_mean | a0_mean]) for a 32-node tile.
__global__ __launch_bounds__(128) void aggr_mlp_kernel(float* __restrict__ hf, bf16* __restrict__ hb,
    const float* __restrict__ a1, const float* __restrict__ a0,
    const float* __restrict__ c1, const float* __restrict__ c0,
    const bf16* W1p, const float* b1, const bf16* W2p, const float* b2,
    const bf16* W3p, const float* b3) {
  __shared__ bf16 X[32 * 384];
  __shared__ bf16 H1s[32 * 256];
  __shared__ bf16 H2s[32 * 128];
  const int t = threadIdx.x;
  const int node0 = blockIdx.x * 32;
  #pragma unroll
  for (int q = t; q < 512; q += 128) {
    const int row = q >> 4;
    const int off = (q & 15) * 8;
    *(v8bf*)(&X[row * 384 + off]) = *(const v8bf*)(hb + (size_t)(node0 + row) * 128 + off);
  }
  #pragma unroll
  for (int q = t; q < 2048; q += 128) {
    const int seg = q >> 10;
    const int rem = q & 1023;
    const int row = rem >> 5;
    const int off = (rem & 31) * 4;
    const int node = node0 + row;
    const float* ap = seg ? a0 : a1;
    const float r = 1.f / fmaxf((seg ? c0 : c1)[node], 1.f);
    float4 v = *(const float4*)(ap + (size_t)node * 128 + off);
    v4bf o; o[0] = (bf16)(v.x * r); o[1] = (bf16)(v.y * r);
    o[2] = (bf16)(v.z * r); o[3] = (bf16)(v.w * r);
    *(v4bf*)(&X[row * 384 + 128 + seg * 128 + off]) = o;
  }
  __syncthreads();
  mlp_layers12<384>(X, H1s, H2s, W1p, b1, W2p, b2);
  const int lane = t & 31, wave = t >> 5;
  const int n = lane & 15, half = lane >> 4;
  v8f c[2][2];
  #pragma unroll
  for (int tt = 0; tt < 2; ++tt) {
    v8f cb = bias_frag(b3, (wave * 2 + tt) * 16);
    c[tt][0] = cb; c[tt][1] = cb;
  }
  gemm_block<128, 8, 2, 2>(H2s, 128, W3p, wave * 2, c);
  #pragma unroll
  for (int tt = 0; tt < 2; ++tt) {
    const int n0 = (wave * 2 + tt) * 16;
    #pragma unroll
    for (int mt = 0; mt < 2; ++mt) {
      #pragma unroll
      for (int j = 0; j < 8; ++j) {
        const int row = node0 + mt * 16 + j + 8 * half;
        const float x = hf[(size_t)row * 128 + n0 + n] + c[tt][mt][j];  // residual
        hf[(size_t)row * 128 + n0 + n] = x;
        hb[(size_t)row * 128 + n0 + n] = (bf16)x;
      }
    }
  }
}

// support = h @ gc_W (128 -> 256); bias applied in the final max kernel.
__global__ __launch_bounds__(128) void gc_support_kernel(const bf16* __restrict__ hb,
    const bf16* __restrict__ gcWp, float* __restrict__ support) {
  const int node0 = blockIdx.x * 32;
  const int lane = threadIdx.x & 31, wave = threadIdx.x >> 5;
  const int n = lane & 15, half = lane >> 4;
  v8f c[4][2] = {};
  gemm_block<128, 16, 4, 2>(hb + (size_t)node0 * 128, 128, gcWp, wave * 4, c);
  #pragma unroll
  for (int tt = 0; tt < 4; ++tt) {
    const int n0 = (wave * 4 + tt) * 16;
    #pragma unroll
    for (int mt = 0; mt < 2; ++mt) {
      #pragma unroll
      for (int j = 0; j < 8; ++j)
        support[(size_t)(node0 + mt * 16 + j + 8 * half) * 256 + n0 + n] = c[tt][mt][j];
    }
  }
}

__global__ __launch_bounds__(256) void gc_scatter_kernel(const float* __restrict__ support,
    const int* __restrict__ src, const int* __restrict__ dst, float* __restrict__ agg) {
  const int e = blockIdx.x;
  const int t = threadIdx.x;
  atomicAdd(&agg[(size_t)dst[e] * 256 + t], support[(size_t)src[e] * 256 + t]);
}

__global__ __launch_bounds__(256) void col_max_kernel(const float* __restrict__ agg,
    const float* __restrict__ gcb, float* __restrict__ out) {
  __shared__ float red[256];
  const int j = blockIdx.x;
  const int t = threadIdx.x;
  float m = -3.402823466e38f;
  for (int nrow = t; nrow < NNODE; nrow += 256) m = fmaxf(m, agg[(size_t)nrow * 256 + j]);
  red[t] = m;
  __syncthreads();
  for (int s = 128; s > 0; s >>= 1) {
    if (t < s) red[t] = fmaxf(red[t], red[t + s]);
    __syncthreads();
  }
  if (t == 0) out[j] = red[0] + gcb[j];
}

// ---------------------------------------------------------------------------
// Host launcher
// ---------------------------------------------------------------------------

extern "C" void kernel_launch(void* const* d_in, const int* in_sizes, int n_in,
                              void* d_out, int out_size, void* d_ws, size_t ws_size,
                              hipStream_t stream) {
  (void)in_sizes; (void)n_in; (void)out_size; (void)ws_size;

  // Inputs in setup_inputs() dict insertion order.
  const float* x_v = (const float*)d_in[0];
  const int*   src = (const int*)d_in[2];
  const int*   dst = (const int*)d_in[3];
  const float* vW1 = (const float*)d_in[4];  const float* vb1 = (const float*)d_in[5];
  const float* vW2 = (const float*)d_in[6];  const float* vb2 = (const float*)d_in[7];
  const float* vW3 = (const float*)d_in[8];  const float* vb3 = (const float*)d_in[9];
  const float* eW1 = (const float*)d_in[10]; const float* eb1 = (const float*)d_in[11];
  const float* eW2 = (const float*)d_in[12]; const float* eb2 = (const float*)d_in[13];
  const float* eW3 = (const float*)d_in[14]; const float* eb3 = (const float*)d_in[15];
  const float* m1W1 = (const float*)d_in[16]; const float* m1b1 = (const float*)d_in[17];
  const float* m1W2 = (const float*)d_in[18]; const float* m1b2 = (const float*)d_in[19];
  const float* m1W3 = (const float*)d_in[20]; const float* m1b3 = (const float*)d_in[21];
  const float* m0W1 = (const float*)d_in[22]; const float* m0b1 = (const float*)d_in[23];
  const float* m0W2 = (const float*)d_in[24]; const float* m0b2 = (const float*)d_in[25];
  const float* m0W3 = (const float*)d_in[26]; const float* m0b3 = (const float*)d_in[27];
  const float* agW1 = (const float*)d_in[28]; const float* agb1 = (const float*)d_in[29];
  const float* agW2 = (const float*)d_in[30]; const float* agb2 = (const float*)d_in[31];
  const float* agW3 = (const float*)d_in[32]; const float* agb3 = (const float*)d_in[33];
  const float* gcW  = (const float*)d_in[34]; const float* gcb  = (const float*)d_in[35];
  float* out = (float*)d_out;

  size_t off = 0;
  auto alloc = [&](size_t bytes) -> char* {
    char* p = (char*)d_ws + off;
    off += (bytes + 255) & ~(size_t)255;
    return p;
  };
  float* hf   = (float*)alloc((size_t)NNODE * 128 * 4);
  bf16*  hb   = (bf16*) alloc((size_t)NNODE * 128 * 2);
  float* a1   = (float*)alloc((size_t)NNODE * 128 * 4);
  float* a0   = (float*)alloc((size_t)NNODE * 128 * 4);
  float* c1   = (float*)alloc((size_t)NNODE * 4);
  float* c0   = (float*)alloc((size_t)NNODE * 4);
  bf16*  he1  = (bf16*) alloc(128 * 2);
  bf16*  he0  = (bf16*) alloc(128 * 2);
  bf16*  bvW1 = (bf16*) alloc(128 * 256 * 2);
  bf16*  bvW2 = (bf16*) alloc(256 * 128 * 2);
  bf16*  bvW3 = (bf16*) alloc(128 * 128 * 2);
  bf16*  b1W1 = (bf16*) alloc(384 * 256 * 2);
  bf16*  b1W2 = (bf16*) alloc(256 * 128 * 2);
  bf16*  b1W3 = (bf16*) alloc(128 * 128 * 2);
  bf16*  b0W1 = (bf16*) alloc(384 * 256 * 2);
  bf16*  b0W2 = (bf16*) alloc(256 * 128 * 2);
  bf16*  b0W3 = (bf16*) alloc(128 * 128 * 2);
  bf16*  bgW1 = (bf16*) alloc(384 * 256 * 2);
  bf16*  bgW2 = (bf16*) alloc(256 * 128 * 2);
  bf16*  bgW3 = (bf16*) alloc(128 * 128 * 2);
  bf16*  bgcW = (bf16*) alloc(128 * 256 * 2);
  float* support = (float*)alloc((size_t)NNODE * 256 * 4);
  float* agg     = (float*)alloc((size_t)NNODE * 256 * 4);

  auto repack = [&](const float* w, bf16* p, int K, int N) {
    repack_weight_kernel<<<(K * N + 255) / 256, 256, 0, stream>>>(w, p, K, N);
  };
  repack(vW1, bvW1, 128, 256);  repack(vW2, bvW2, 256, 128);  repack(vW3, bvW3, 128, 128);
  repack(m1W1, b1W1, 384, 256); repack(m1W2, b1W2, 256, 128); repack(m1W3, b1W3, 128, 128);
  repack(m0W1, b0W1, 384, 256); repack(m0W2, b0W2, 256, 128); repack(m0W3, b0W3, 128, 128);
  repack(agW1, bgW1, 384, 256); repack(agW2, bgW2, 256, 128); repack(agW3, bgW3, 128, 128);
  repack(gcW, bgcW, 128, 256);

  he_kernel<<<1, 256, 0, stream>>>(eW1, eb1, eW2, eb2, eW3, eb3, he1, he0);

  node_mlp_kernel<<<NNODE / 32, 128, 0, stream>>>(x_v, bvW1, vb1, bvW2, vb2, bvW3, vb3, hf, hb);

  hipMemsetAsync(c1, 0, (size_t)NNODE * 4, stream);
  hipMemsetAsync(c0, 0, (size_t)NNODE * 4, stream);
  count_kernel<<<(NEDGE + 255) / 256, 256, 0, stream>>>(dst, c1, c0);

  for (int it = 0; it < TITER; ++it) {
    hipMemsetAsync(a1, 0, (size_t)NNODE * 128 * 4, stream);
    hipMemsetAsync(a0, 0, (size_t)NNODE * 128 * 4, stream);
    edge_mlp_kernel<<<NEDGE1 / 32, 128, 0, stream>>>(hb, src, dst, he1,
        b1W1, m1b1, b1W2, m1b2, b1W3, m1b3, a1);
    edge_mlp_kernel<<<(NEDGE - NEDGE1) / 32, 128, 0, stream>>>(hb, src + NEDGE1, dst + NEDGE1, he0,
        b0W1, m0b1, b0W2, m0b2, b0W3, m0b3, a0);
    aggr_mlp_kernel<<<NNODE / 32, 128, 0, stream>>>(hf, hb, a1, a0, c1, c0,
        bgW1, agb1, bgW2, agb2, bgW3, agb3);
  }

  gc_support_kernel<<<NNODE / 32, 128, 0, stream>>>(hb, bgcW, support);
  hipMemsetAsync(agg, 0, (size_t)NNODE * 256 * 4, stream);
  gc_scatter_kernel<<<NEDGE, 256, 0, stream>>>(support, src, dst, agg);
  col_max_kernel<<<256, 256, 0, stream>>>(agg, gcb, out);
}
